// EncoderLayer_48533130445179
// MI455X (gfx1250) — matrix operand order repaired and checked by reading, not compiled, not run on hardware
//
#include <hip/hip_runtime.h>
#include <hip/hip_bf16.h>
#include <stdint.h>
#include <math.h>

typedef unsigned short u16;
typedef unsigned int u32;
typedef __attribute__((ext_vector_type(16))) __bf16 bf16x16;
typedef __attribute__((ext_vector_type(8))) float v8f;
typedef int v4i_vs __attribute__((vector_size(16)));   // matches async-LDS builtin param

#define AS1 __attribute__((address_space(1)))
#define AS3 __attribute__((address_space(3)))

union FragAB { bf16x16 v; u32 u[8]; };
union FragC  { v8f v; float f[8]; };

__device__ inline u16 f2bf(float f) {
  u32 x; __builtin_memcpy(&x, &f, 4);
  x += 0x7fffu + ((x >> 16) & 1u);          // round-to-nearest-even
  return (u16)(x >> 16);
}

// A-fragment K index for VGPR i, lane-half hi (16-bit A 16x32 layout, ISA 7.12.2)
__device__ inline int kmapA(int i, int hi) {
  return (i < 4) ? (hi * 8 + 2 * i) : (16 + hi * 8 + 2 * (i - 4));
}

// Fused DPP16 ROW_XMASK max: one VOP2+DPP instruction per butterfly step
#define DPP_MAX(NAME, MASK)                                                     \
  __device__ inline float NAME(float x) {                                       \
    float o;                                                                    \
    asm("v_max_num_f32_dpp %0, %1, %1 row_xmask:" #MASK                         \
        " row_mask:0xf bank_mask:0xf bound_ctrl:1"                              \
        : "=v"(o) : "v"(x));                                                    \
    return o;                                                                   \
  }
DPP_MAX(dpp_max1, 1)
DPP_MAX(dpp_max2, 2)
DPP_MAX(dpp_max4, 4)
DPP_MAX(dpp_max8, 8)

// Async global->LDS 16B copy (ASYNCcnt-tracked, bypasses VGPRs)
__device__ inline void async_copy16(const void* gsrc, void* ldst) {
#if __has_builtin(__builtin_amdgcn_global_load_async_to_lds_b128)
  __builtin_amdgcn_global_load_async_to_lds_b128(
      (AS1 v4i_vs*)gsrc, (AS3 v4i_vs*)ldst, 0, 0);
#else
  u32 loff = (u32)(uintptr_t)(AS3 void*)ldst;
  asm volatile("global_load_async_to_lds_b128 %0, %1, off"
               :: "v"(loff), "v"(gsrc) : "memory");
#endif
}

__device__ inline void wait_async0() {
#if __has_builtin(__builtin_amdgcn_s_wait_asynccnt)
  __builtin_amdgcn_s_wait_asynccnt(0);
#else
  asm volatile("s_wait_asynccnt 0" ::: "memory");
#endif
}

// ---------------------------------------------------------------------------
// fp32 -> bf16 weight conversion (packed 2-per-thread)
// ---------------------------------------------------------------------------
__global__ __launch_bounds__(256) void cvt_bf16(const float* __restrict__ in,
                                                u16* __restrict__ outp, int n2) {
  int i = blockIdx.x * blockDim.x + threadIdx.x;
  if (i < n2) {
    float a = in[2 * i], b = in[2 * i + 1];
    u32 p = (u32)f2bf(a) | ((u32)f2bf(b) << 16);
    *(u32*)(outp + 2 * i) = p;
  }
}

// ---------------------------------------------------------------------------
// (optional residual add) + LayerNorm -> bf16 activations (+ fp32 x2 copy)
// ---------------------------------------------------------------------------
__global__ __launch_bounds__(256) void ln_fuse(
    const float* __restrict__ xin, const float* __restrict__ resid,
    float* __restrict__ x2out, const float* __restrict__ g,
    const float* __restrict__ bta, u16* __restrict__ xn) {
  __shared__ float rs[256], rq[256];
  const int tok = blockIdx.x, t = threadIdx.x;
  const float* xp = xin + (size_t)tok * 512;
  float v0 = xp[t], v1 = xp[t + 256];
  if (resid) {
    const float* rp = resid + (size_t)tok * 512;
    v0 += rp[t]; v1 += rp[t + 256];
  }
  if (x2out) {
    float* op = x2out + (size_t)tok * 512;
    op[t] = v0; op[t + 256] = v1;
  }
  rs[t] = v0 + v1;
  rq[t] = v0 * v0 + v1 * v1;
  __syncthreads();
  for (int s = 128; s > 0; s >>= 1) {
    if (t < s) { rs[t] += rs[t + s]; rq[t] += rq[t + s]; }
    __syncthreads();
  }
  float mean = rs[0] * (1.0f / 512.0f);
  float var  = rq[0] * (1.0f / 512.0f) - mean * mean;
  float rinv = rsqrtf(var + 1e-5f);
  u16* op = xn + (size_t)tok * 512;
  op[t]       = f2bf((v0 - mean) * rinv * g[t] + bta[t]);
  op[t + 256] = f2bf((v1 - mean) * rinv * g[t + 256] + bta[t + 256]);
}

// ---------------------------------------------------------------------------
// WMMA bf16 GEMM: C[M,Nout] = A[M,K] * W[K,Nout] (+bias, templated epilogue)
// block tile 128x128, BK=32, 8 waves (2x4), wave tile 64x32 (4x2 wmma tiles)
// A tile staged with async global->LDS copies; B tile transposed via VGPRs.
// ---------------------------------------------------------------------------
template <int EPI>
__global__ __launch_bounds__(256) void gemm_bf16(
    const u16* __restrict__ A, const u16* __restrict__ W,
    const float* __restrict__ bias, int M, int K, int Nout,
    u16* __restrict__ Qb, u16* __restrict__ Kb, u16* __restrict__ Vt,
    u16* __restrict__ hid, const float* __restrict__ res,
    float* __restrict__ out) {
  __shared__ u16 sA[128 * 40];   // row-major, stride 40 (bank spread, 16B-aligned rows)
  __shared__ u16 sB[128 * 40];   // col-major [n][k], stride 40

  const int t = threadIdx.x;
  const int lane = t & 31, w = t >> 5;
  const int wm = w >> 2, wn = w & 3;
  const int hi = lane >> 4, lr = lane & 15;
  const int m0 = blockIdx.y * 128, n0 = blockIdx.x * 128;

  FragC acc[4][2];
#pragma unroll
  for (int a = 0; a < 4; ++a)
#pragma unroll
    for (int b = 0; b < 2; ++b)
#pragma unroll
      for (int r = 0; r < 8; ++r) acc[a][b].f[r] = 0.f;

  for (int kk = 0; kk < K; kk += 32) {
    {  // stage A tile 128x32 via async copies (16B per lane, ASYNCcnt)
      int r = t >> 2, c8 = (t & 3) * 8;
#pragma unroll
      for (int it = 0; it < 2; ++it, r += 64)
        async_copy16(A + (size_t)(m0 + r) * K + kk + c8, &sA[r * 40 + c8]);
    }
    {  // stage B tile 32x128 transposed into col-major LDS
      int k = t >> 4, n8 = (t & 15) * 8;
#pragma unroll
      for (int it = 0; it < 2; ++it, k += 16) {
        uint4 val = *(const uint4*)(W + (size_t)(kk + k) * Nout + n0 + n8);
        u32 e[4] = {val.x, val.y, val.z, val.w};
#pragma unroll
        for (int j = 0; j < 4; ++j) {
          sB[(n8 + 2 * j) * 40 + k]     = (u16)(e[j] & 0xffff);
          sB[(n8 + 2 * j + 1) * 40 + k] = (u16)(e[j] >> 16);
        }
      }
    }
    wait_async0();
    __syncthreads();

    FragAB af[4], bfr[2];
#pragma unroll
    for (int b = 0; b < 2; ++b) {
      int col = wn * 32 + b * 16 + lr;
#pragma unroll
      for (int i = 0; i < 8; ++i)
        bfr[b].u[i] = *(const u32*)(sB + col * 40 + kmapA(i, hi));
    }
#pragma unroll
    for (int a = 0; a < 4; ++a) {
      int row = wm * 64 + a * 16 + lr;
#pragma unroll
      for (int i = 0; i < 8; ++i)
        af[a].u[i] = *(const u32*)(sA + row * 40 + kmapA(i, hi));
    }
#pragma unroll
    for (int a = 0; a < 4; ++a)
#pragma unroll
      for (int b = 0; b < 2; ++b)
        acc[a][b].v = __builtin_amdgcn_wmma_f32_16x16x32_bf16(
            false, af[a].v, false, bfr[b].v, (short)0, acc[a][b].v, false, false);
    __syncthreads();
  }

  // epilogue
#pragma unroll
  for (int a = 0; a < 4; ++a)
#pragma unroll
    for (int b = 0; b < 2; ++b)
#pragma unroll
      for (int r = 0; r < 8; ++r) {
        int rg = m0 + wm * 64 + a * 16 + hi * 8 + r;
        int cg = n0 + wn * 32 + b * 16 + lr;
        float val = acc[a][b].f[r] + bias[cg];
        if (EPI == 0) {
          // qkv column layout: reshape(3E) -> (H, 3*HD), split thirds of 3*HD
          int bidx = rg >> 11, tok = rg & 2047;
          int h = cg / 192, rem = cg % 192;
          int sel = rem >> 6, d = rem & 63;
          int bh = bidx * 8 + h;
          if (sel == 0)
            Qb[((size_t)bh * 2048 + tok) * 64 + d] = f2bf(val * 0.125f);  // HD^-0.5
          else if (sel == 1)
            Kb[((size_t)bh * 2048 + tok) * 64 + d] = f2bf(val);
          else
            Vt[((size_t)bh * 64 + d) * 2048 + tok] = f2bf(val);
        } else if (EPI == 1) {
          float gl = 0.5f * val * (1.0f + erff(val * 0.70710678118654752f));
          hid[(size_t)rg * 1536 + cg] = f2bf(gl);
        } else {
          out[(size_t)rg * 512 + cg] = val + res[(size_t)rg * 512 + cg];
        }
      }
}

// ---------------------------------------------------------------------------
// Flash attention: 1 wave per (b,h, 16-row q tile); online softmax over N=2048
// keys, 32 keys / iter: 4 score WMMA + 1 rowsum WMMA + 4 PV WMMA.
// Row max via fused DPP ROW_XMASK max; row sum via ones-matrix WMMA.
// ---------------------------------------------------------------------------
__global__ __launch_bounds__(32) void attn_fa(
    const u16* __restrict__ Qb, const u16* __restrict__ Kb,
    const u16* __restrict__ Vt, float* __restrict__ ob) {
  __shared__ u16 sP[16 * 36];  // P tile 16x32 bf16, padded stride
  const int lane = threadIdx.x;
  const int hi = lane >> 4, lr = lane & 15;
  const int qt = blockIdx.x & 127, bh = blockIdx.x >> 7;
  const u16* Qp = Qb + ((size_t)bh * 2048 + qt * 16) * 64;
  const u16* Kp = Kb + (size_t)bh * 2048 * 64;
  const u16* Vp = Vt + (size_t)bh * 64 * 2048;

  FragAB qf[2];
#pragma unroll
  for (int j = 0; j < 2; ++j)
#pragma unroll
    for (int i = 0; i < 8; ++i)
      qf[j].u[i] = *(const u32*)(Qp + lr * 64 + j * 32 + kmapA(i, hi));

  FragAB onesB;  // all-1.0 bf16 B fragment for row-sum WMMA
#pragma unroll
  for (int i = 0; i < 8; ++i) onesB.u[i] = 0x3f803f80u;

  FragC o[4];
  float mrow[8], lrow[8], alpha[8];
#pragma unroll
  for (int r = 0; r < 8; ++r) { mrow[r] = -1e30f; lrow[r] = 0.f; }
#pragma unroll
  for (int j = 0; j < 4; ++j)
#pragma unroll
    for (int r = 0; r < 8; ++r) o[j].f[r] = 0.f;

  for (int key0 = 0; key0 < 2048; key0 += 32) {
    FragC s0, s1;
#pragma unroll
    for (int r = 0; r < 8; ++r) { s0.f[r] = 0.f; s1.f[r] = 0.f; }
    FragAB kfr;
#pragma unroll
    for (int j = 0; j < 2; ++j) {  // two K-steps over HD=64
#pragma unroll
      for (int i = 0; i < 8; ++i)
        kfr.u[i] = *(const u32*)(Kp + (size_t)(key0 + lr) * 64 + j * 32 + kmapA(i, hi));
      s0.v = __builtin_amdgcn_wmma_f32_16x16x32_bf16(false, qf[j].v, false, kfr.v,
                                                     (short)0, s0.v, false, false);
    }
#pragma unroll
    for (int j = 0; j < 2; ++j) {
#pragma unroll
      for (int i = 0; i < 8; ++i)
        kfr.u[i] = *(const u32*)(Kp + (size_t)(key0 + 16 + lr) * 64 + j * 32 + kmapA(i, hi));
      s1.v = __builtin_amdgcn_wmma_f32_16x16x32_bf16(false, qf[j].v, false, kfr.v,
                                                     (short)0, s1.v, false, false);
    }
    // online softmax: row max via fused DPP xor-butterfly within 16-lane halves
#pragma unroll
    for (int r = 0; r < 8; ++r) {
      float a = fmaxf(s0.f[r], s1.f[r]);
      a = dpp_max1(a);
      a = dpp_max2(a);
      a = dpp_max4(a);
      a = dpp_max8(a);
      float mn = fmaxf(mrow[r], a);
      alpha[r] = __expf(mrow[r] - mn);
      mrow[r] = mn;
      float p0 = __expf(s0.f[r] - mn);
      float p1 = __expf(s1.f[r] - mn);
      sP[(hi * 8 + r) * 36 + lr]      = f2bf(p0);
      sP[(hi * 8 + r) * 36 + 16 + lr] = f2bf(p1);
    }
#pragma unroll
    for (int j = 0; j < 4; ++j)
#pragma unroll
      for (int r = 0; r < 8; ++r) o[j].f[r] *= alpha[r];
    __syncthreads();
    FragAB pf;
#pragma unroll
    for (int i = 0; i < 8; ++i)
      pf.u[i] = *(const u32*)(sP + lr * 36 + kmapA(i, hi));
    // row sums of P via ones-matrix WMMA (replicated across columns)
    FragC ps;
#pragma unroll
    for (int r = 0; r < 8; ++r) ps.f[r] = 0.f;
    ps.v = __builtin_amdgcn_wmma_f32_16x16x32_bf16(false, pf.v, false, onesB.v,
                                                   (short)0, ps.v, false, false);
#pragma unroll
    for (int r = 0; r < 8; ++r) lrow[r] = lrow[r] * alpha[r] + ps.f[r];
#pragma unroll
    for (int j = 0; j < 4; ++j) {
      FragAB vb;
#pragma unroll
      for (int i = 0; i < 8; ++i)
        vb.u[i] = *(const u32*)(Vp + (size_t)(j * 16 + lr) * 2048 + key0 + kmapA(i, hi));
      o[j].v = __builtin_amdgcn_wmma_f32_16x16x32_bf16(false, pf.v, false, vb.v,
                                                       (short)0, o[j].v, false, false);
    }
    __syncthreads();
  }
  const int b = bh >> 3, h = bh & 7;
#pragma unroll
  for (int j = 0; j < 4; ++j)
#pragma unroll
    for (int r = 0; r < 8; ++r) {
      int tok = qt * 16 + hi * 8 + r;
      int d = j * 16 + lr;
      ob[((size_t)b * 2048 + tok) * 512 + h * 64 + d] = o[j].f[r] / lrow[r];
    }
}

// ---------------------------------------------------------------------------
extern "C" void kernel_launch(void* const* d_in, const int* in_sizes, int n_in,
                              void* d_out, int out_size, void* d_ws, size_t ws_size,
                              hipStream_t stream) {
  (void)in_sizes; (void)n_in; (void)out_size; (void)ws_size;
  const float* x     = (const float*)d_in[0];
  const float* qkv_w = (const float*)d_in[1];
  const float* qkv_b = (const float*)d_in[2];
  const float* fc1_w = (const float*)d_in[3];
  const float* fc1_b = (const float*)d_in[4];
  const float* fc2_w = (const float*)d_in[5];
  const float* fc2_b = (const float*)d_in[6];
  const float* lag   = (const float*)d_in[7];
  const float* lab   = (const float*)d_in[8];
  const float* lfg   = (const float*)d_in[9];
  const float* lfb   = (const float*)d_in[10];
  float* out = (float*)d_out;

  const int M = 4 * 2048, E = 512, HID = 1536;
  char* p = (char*)d_ws;
  auto alloc = [&](size_t bytes) -> char* {
    char* r = p;
    p += (bytes + 255) & ~(size_t)255;
    return r;
  };
  u16*   xn1  = (u16*)alloc((size_t)M * E * 2);
  u16*   wqkv = (u16*)alloc((size_t)E * HID * 2);
  u16*   wfc1 = (u16*)alloc((size_t)E * HID * 2);
  u16*   wfc2 = (u16*)alloc((size_t)HID * E * 2);
  u16*   Qbuf = (u16*)alloc((size_t)M * E * 2);
  u16*   Kbuf = (u16*)alloc((size_t)M * E * 2);
  u16*   Vtb  = (u16*)alloc((size_t)M * E * 2);
  float* ab   = (float*)alloc((size_t)M * E * 4);
  float* x2   = (float*)alloc((size_t)M * E * 4);
  u16*   xn2  = (u16*)alloc((size_t)M * E * 2);
  u16*   hid  = (u16*)alloc((size_t)M * HID * 2);

  int n2 = E * HID / 2;
  cvt_bf16<<<(n2 + 255) / 256, 256, 0, stream>>>(qkv_w, wqkv, n2);
  cvt_bf16<<<(n2 + 255) / 256, 256, 0, stream>>>(fc1_w, wfc1, n2);
  cvt_bf16<<<(n2 + 255) / 256, 256, 0, stream>>>(fc2_w, wfc2, n2);

  // pre-norm 1
  ln_fuse<<<M, 256, 0, stream>>>(x, nullptr, nullptr, lag, lab, xn1);
  // QKV projection + scatter
  gemm_bf16<0><<<dim3(HID / 128, M / 128), 256, 0, stream>>>(
      xn1, wqkv, qkv_b, M, E, HID, Qbuf, Kbuf, Vtb, nullptr, nullptr, nullptr);
  // attention (B*H*N/16 = 4096 wave-blocks)
  attn_fa<<<4096, 32, 0, stream>>>(Qbuf, Kbuf, Vtb, ab);
  // residual + pre-norm 2
  ln_fuse<<<M, 256, 0, stream>>>(x, ab, x2, lfg, lfb, xn2);
  // FFN
  gemm_bf16<1><<<dim3(HID / 128, M / 128), 256, 0, stream>>>(
      xn2, wfc1, fc1_b, M, E, HID, nullptr, nullptr, nullptr, hid, nullptr, nullptr);
  gemm_bf16<2><<<dim3(E / 128, M / 128), 256, 0, stream>>>(
      hid, wfc2, fc2_b, M, HID, E, nullptr, nullptr, nullptr, nullptr, x2, out);
}
